// Encode_27169963114665
// MI455X (gfx1250) — compile-verified, matrix operand order checked by reading, not run on hardware
//
#include <hip/hip_runtime.h>
#include <hip/hip_bf16.h>
#include <cmath>

typedef __attribute__((ext_vector_type(16))) _Float16 v16h_t;
typedef __attribute__((ext_vector_type(8)))  float    v8f_t;
typedef __attribute__((ext_vector_type(4)))  unsigned int u32x4_t;
typedef __attribute__((ext_vector_type(8)))  unsigned int u32x8_t;

#define NTHR 256

// ---- f16 weight scratch layout in d_ws (element offsets) ----
#define OFF_W1   0          // 64  x 96
#define OFF_WR1  6144       // 64  x 32
#define OFF_WD1  8192       // 64  x 192
#define OFF_WR2  20480      // 128 x 64
#define OFF_W2   28672      // 128 x 192
#define OFF_WD2  53248      // 32  x 384
#define OFF_WR3  65536      // 32  x 128
#define OFF_WF   69632      // 32  x 32
#define W_TOTAL  70656
#define FUSED_BYTE_OFF (W_TOTAL * 2)   // then 64x32 f32 "fused" buffer

// -------------------------------------------------------------------------
// Weight repack: f32 (rows x Kreal) -> f16 (rows x Kpad), zero padded.
// -------------------------------------------------------------------------
__global__ void prep_weights(const float* __restrict__ src, _Float16* __restrict__ dst,
                             int rows, int realRows, int Kreal, int Kpad) {
  int e = blockIdx.x * 256 + threadIdx.x;
  if (e >= rows * Kpad) return;
  int r = e / Kpad, k = e - r * Kpad;
  float v = (r < realRows && k < Kreal) ? src[r * Kreal + k] : 0.f;
  dst[e] = (_Float16)v;
}

// -------------------------------------------------------------------------
// WMMA fragment load: 16x32 f16 tile, row-major source with stride ld.
// Per ISA 7.12.2 (16-bit A 16x32): lane = {m | hsel<<4}; VGPR v holds K pair
// at (v>>2)*16 + hsel*8 + (v&3)*2. B is read from B^T with identical layout.
// -------------------------------------------------------------------------
__device__ __forceinline__ v16h_t load_frag(const _Float16* __restrict__ p, int hsel) {
  v16h_t a;
#pragma unroll
  for (int u = 0; u < 8; ++u) {
    int kb = ((u >> 2) << 4) + (hsel << 3) + ((u & 3) << 1);
    a[2 * u]     = p[kb];
    a[2 * u + 1] = p[kb + 1];
  }
  return a;
}

// -------------------------------------------------------------------------
// Block-cooperative GEMM: C(MxN) = A(MxK) * B(KxN), A = f16 weights (global,
// row-major, lda=Kpad), B given as B^T (NxK f16 in LDS, ldb=Kpad).
// M%16==0, N%16==0, K%32==0. One 16x16 tile per wave iteration; epilogue
// lambda handles bias/BN/ReLU/residual and the store.
// -------------------------------------------------------------------------
template <typename Epi>
__device__ __forceinline__ void gemm_wmma(const _Float16* __restrict__ A, int lda,
                                          const _Float16* __restrict__ Bt, int ldb,
                                          int M, int N, int K, Epi epi) {
  const int tid  = threadIdx.x;
  const int lane = tid & 31;
  const int wave = tid >> 5;
  const int nwv  = NTHR >> 5;
  const int lidx = lane & 15;
  const int hsel = lane >> 4;
  const int nt   = N >> 4;
  const int tiles = (M >> 4) * nt;
  for (int t = wave; t < tiles; t += nwv) {
    const int mt = t / nt;
    const int m0 = mt << 4;
    const int n0 = (t - mt * nt) << 4;
    v8f_t acc = {};
    for (int k0 = 0; k0 < K; k0 += 32) {
      v16h_t a = load_frag(A  + (size_t)(m0 + lidx) * lda + k0, hsel);
      v16h_t b = load_frag(Bt + (size_t)(n0 + lidx) * ldb + k0, hsel);
      acc = __builtin_amdgcn_wmma_f32_16x16x32_f16(
          /*neg_a=*/false, a, /*neg_b=*/false, b,
          /*c_mod=*/(short)0, acc, /*reuse_a=*/false, /*reuse_b=*/false);
    }
    const int n  = n0 + lidx;
    const int mb = m0 + (hsel << 3);
#pragma unroll
    for (int u = 0; u < 8; ++u) epi(mb + u, n, acc[u]);
  }
}

// -------------------------------------------------------------------------
// B^T builders (im2col into LDS, f16). Kpad columns, zero padded.
// -------------------------------------------------------------------------
__device__ __forceinline__ void build_bt_conv3(_Float16* Bt, const float* src, int Cin,
                                               int Lin, int Lout, int stride, int Kpad) {
  int total = Lout * Kpad;
  for (int e = threadIdx.x; e < total; e += NTHR) {
    int l = e / Kpad, kk = e - l * Kpad;
    float val = 0.f;
    if (kk < Cin * 3) {
      int c = kk / 3, t = kk - c * 3;
      int pos = l * stride + t - 1;            // pad = 1
      if (pos >= 0 && pos < Lin) val = src[c * Lin + pos];
    }
    Bt[e] = (_Float16)val;
  }
}

__device__ __forceinline__ void build_bt_1x1(_Float16* Bt, const float* src,
                                             int Cin, int L, int Kpad) {
  int total = L * Kpad;
  for (int e = threadIdx.x; e < total; e += NTHR) {
    int l = e / Kpad, c = e - l * Kpad;
    Bt[e] = (_Float16)((c < Cin) ? src[c * L + l] : 0.f);
  }
}

__device__ __forceinline__ void build_bt_1x1_avg(_Float16* Bt, const float* src,
                                                 int Cin, int Lout, int Kpad) {
  int total = Lout * Kpad;
  for (int e = threadIdx.x; e < total; e += NTHR) {
    int l = e / Kpad, c = e - l * Kpad;
    float v = 0.f;
    if (c < Cin) v = 0.5f * (src[c * 2 * Lout + 2 * l] + src[c * 2 * Lout + 2 * l + 1]);
    Bt[e] = (_Float16)v;
  }
}

// -------------------------------------------------------------------------
// Fused encoder: one block = one batch item (of 64). Everything in LDS.
// X tile (22x64 f32) is brought in by the Tensor Data Mover as a single 2D
// descriptor DMA issued by wave 0; completion via s_wait_tensorcnt.
// -------------------------------------------------------------------------
__global__ __launch_bounds__(NTHR)
void encode_kernel(const float* __restrict__ x, const _Float16* __restrict__ wf16,
                   float* __restrict__ fusedG,
                   const float* b1, const float* g1, const float* bt1,
                   const float* m1, const float* v1,
                   const float* br1, const float* bd1,
                   const float* b2, const float* g2, const float* bt2,
                   const float* m2, const float* v2,
                   const float* br2, const float* bd2, const float* br3,
                   const float* bfb, const float* ca_w1, const float* ca_w2) {
  __shared__ float    sX[2048];     // X(22x64) -> hd(64x32) -> hw(32x16)
  __shared__ float    sA1[4096];    // h(64x64) -> h2(128x32) -> h3(32x16) -> hf(22x32)
  __shared__ float    sA2[4096];    // idn(64x64) -> idn2(128x32) -> interp(32x32)
  __shared__ _Float16 sBt[6144];    // B^T staging (max 12KB)
  __shared__ float    sCAavg[22], sCAmx[22], sCAh[22], sGate[22];

  const int tid  = threadIdx.x;
  const int item = blockIdx.x;          // 0..63 ; item = n*8 + seg
  const int nidx = item >> 3;
  const int seg  = item & 7;
  const int wave = tid >> 5;

  // warm L2 with the f16 weight pack (global_prefetch_b8)
  {
    const char* p = (const char*)wf16;
    for (int off = tid * 64; off < W_TOTAL * 2; off += NTHR * 64)
      __builtin_prefetch(p + off, 0, 1);
  }

  // ---- TDM: xi[item][c][l] = x[n][c][seg*64 + l] -> sX (22 rows x 64 f32) ----
  // D# per CDNA5 ISA 8.3/8.4: 2D tile 64x22, data_size=4B, tensor row stride 512,
  // tile rows packed contiguously into LDS (256B/row == sX row stride).
  if (wave == 0) {
    unsigned long long ga =
        (unsigned long long)(const void*)(x + ((size_t)nidx * 22) * 512 + (size_t)seg * 64);
    unsigned lds_addr = (unsigned)(unsigned long long)(&sX[0]);  // low 32 = LDS offset
    u32x4_t g0;
    g0[0] = 1u;                                               // count=1, user descriptor
    g0[1] = lds_addr;                                         // lds_addr
    g0[2] = (unsigned)ga;                                     // global_addr[31:0]
    g0[3] = ((unsigned)(ga >> 32) & 0x01FFFFFFu) | 0x80000000u; // addr[56:32] | type=2
    u32x8_t g1;
    g1[0] = 0x00020000u;          // data_size=2 (4B); no multicast/iterate/pad
    g1[1] = 512u << 16;           // tensor_dim0 = 512 (low 16 in [63:48])
    g1[2] = 22u << 16;            // tensor_dim0 hi=0 ; tensor_dim1 = 22
    g1[3] = 64u << 16;            // tensor_dim1 hi=0 ; tile_dim0 = 64
    g1[4] = 22u;                  // tile_dim1 = 22 ; tile_dim2 = 0
    g1[5] = 512u;                 // tensor_dim0_stride[31:0] = 512
    g1[6] = 0u;                   // stride hi / dim1_stride lo
    g1[7] = 0u;                   // dim1_stride hi
    asm volatile("tensor_load_to_lds %0, %1" :: "s"(g0), "s"(g1) : "memory");
    __builtin_amdgcn_s_wait_tensorcnt(0);
  }
  __syncthreads();

  // ---- idn = conv1x1(wr1, X) : (64,64), K=22->32 ----
  build_bt_1x1(sBt, sX, 22, 64, 32);
  __syncthreads();
  gemm_wmma(wf16 + OFF_WR1, 32, sBt, 32, 64, 64, 32,
            [&](int m, int n, float val) { sA2[m * 64 + n] = val + br1[m]; });
  __syncthreads();

  // ---- h = relu(relu(bn1(conv1(X))) + idn) : (64,64), K=66->96 ----
  build_bt_conv3(sBt, sX, 22, 64, 64, 1, 96);
  __syncthreads();
  gemm_wmma(wf16 + OFF_W1, 96, sBt, 96, 64, 64, 96,
            [&](int m, int n, float val) {
              val += b1[m];
              float sc = g1[m] * rsqrtf(v1[m] + 1e-5f);
              val = (val - m1[m]) * sc + bt1[m];
              val = fmaxf(val, 0.f) + sA2[m * 64 + n];
              sA1[m * 64 + n] = fmaxf(val, 0.f);
            });
  __syncthreads();

  // ---- hd = conv(wd1, h, stride2) : (64,32), K=192 ----
  build_bt_conv3(sBt, sA1, 64, 64, 32, 2, 192);
  __syncthreads();
  gemm_wmma(wf16 + OFF_WD1, 192, sBt, 192, 64, 32, 192,
            [&](int m, int n, float val) { sX[m * 32 + n] = val + bd1[m]; });
  __syncthreads();

  // ---- idn2 = conv1x1(wr2, avgpool2(idn)) : (128,32), K=64 ----
  build_bt_1x1_avg(sBt, sA2, 64, 32, 64);
  __syncthreads();
  gemm_wmma(wf16 + OFF_WR2, 64, sBt, 64, 128, 32, 64,
            [&](int m, int n, float val) { sA2[m * 32 + n] = val + br2[m]; });
  __syncthreads();

  // ---- h2 = relu(relu(bn2(conv2(hd))) + idn2) : (128,32), K=192 ----
  build_bt_conv3(sBt, sX, 64, 32, 32, 1, 192);
  __syncthreads();
  gemm_wmma(wf16 + OFF_W2, 192, sBt, 192, 128, 32, 192,
            [&](int m, int n, float val) {
              val += b2[m];
              float sc = g2[m] * rsqrtf(v2[m] + 1e-5f);
              val = (val - m2[m]) * sc + bt2[m];
              val = fmaxf(val, 0.f) + sA2[m * 32 + n];
              sA1[m * 32 + n] = fmaxf(val, 0.f);
            });
  __syncthreads();

  // ---- hw = conv(wd2, h2, stride2) : (32,16), K=384 ----
  build_bt_conv3(sBt, sA1, 128, 32, 16, 2, 384);
  __syncthreads();
  gemm_wmma(wf16 + OFF_WD2, 384, sBt, 384, 32, 16, 384,
            [&](int m, int n, float val) { sX[m * 16 + n] = val + bd2[m]; });
  __syncthreads();

  // ---- h3 = relu(hw + conv1x1(wr3, avgpool2(idn2))) : (32,16), K=128 ----
  build_bt_1x1_avg(sBt, sA2, 128, 16, 128);
  __syncthreads();
  gemm_wmma(wf16 + OFF_WR3, 128, sBt, 128, 32, 16, 128,
            [&](int m, int n, float val) {
              sA1[m * 16 + n] = fmaxf(sX[m * 16 + n] + val + br3[m], 0.f);
            });
  __syncthreads();

  // ---- linear interp 16 -> 32 (align_corners) into sA2 (32x32) ----
  for (int e = tid; e < 32 * 32; e += NTHR) {
    int c = e >> 5, j = e & 31;
    float pos = (float)j * 15.f / 31.f;
    int lo = (int)floorf(pos);
    int hi = (lo + 1 < 15) ? lo + 1 : 15;
    float w = pos - (float)lo;
    sA2[c * 32 + j] = sA1[c * 16 + lo] * (1.f - w) + sA1[c * 16 + hi] * w;
  }
  __syncthreads();

  // ---- hf = conv1x1(wf, interp) : (22,32), M padded to 32, K=32 ----
  build_bt_1x1(sBt, sA2, 32, 32, 32);
  __syncthreads();
  gemm_wmma(wf16 + OFF_WF, 32, sBt, 32, 32, 32, 32,
            [&](int m, int n, float val) {
              if (m < 22) sA1[m * 32 + n] = val + bfb[m];
            });
  __syncthreads();

  // ---- channel attention ----
  if (tid < 22) {
    float s = 0.f, mx = -INFINITY;
    for (int l = 0; l < 32; ++l) {
      float v = sA1[tid * 32 + l];
      s += v;
      mx = fmaxf(mx, v);
    }
    sCAavg[tid] = s * (1.f / 32.f);
    sCAmx[tid]  = mx;
  }
  __syncthreads();
  if (tid < 22) {                         // hidden = relu(v @ ca_w1^T), 2 x 11
    int j = tid % 11, which = tid / 11;
    const float* vsrc = which ? sCAmx : sCAavg;
    float h = 0.f;
    for (int c = 0; c < 22; ++c) h += vsrc[c] * ca_w1[j * 22 + c];
    sCAh[tid] = fmaxf(h, 0.f);
  }
  __syncthreads();
  if (tid < 22) {                         // gate = sigmoid((h_avg + h_mx) @ ca_w2^T)
    float g = 0.f;
    for (int j = 0; j < 11; ++j) g += (sCAh[j] + sCAh[11 + j]) * ca_w2[tid * 11 + j];
    sGate[tid] = 1.f / (1.f + expf(-g));
  }
  __syncthreads();
  if (tid < 32) {                         // fused[l] = sum_c hf[c][l] * gate[c]
    float f = 0.f;
    for (int c = 0; c < 22; ++c) f += sA1[c * 32 + tid] * sGate[c];
    fusedG[item * 32 + tid] = f;
  }
}

// -------------------------------------------------------------------------
// Self-attention tail: one block per outer batch (8), trivial sizes.
// -------------------------------------------------------------------------
__global__ __launch_bounds__(64)
void attn_kernel(const float* __restrict__ fusedG,
                 const float* wq, const float* bq, const float* wk, const float* bk,
                 const float* wv, const float* bv, const float* wo, const float* bo,
                 float* __restrict__ out) {
  __shared__ float xo[8 * 32], Q[8 * 64], Km[8 * 64], Vm[8 * 64];
  __shared__ float attn[8 * 8], pooled[64];
  const int b = blockIdx.x, tid = threadIdx.x;

  for (int e = tid; e < 256; e += 64) xo[e] = fusedG[b * 256 + e];
  __syncthreads();

  for (int e = tid; e < 512; e += 64) {       // Q,K,V projections
    int r = e >> 6, d = e & 63;
    float q = bq[d], k = bk[d], v = bv[d];
    for (int c = 0; c < 32; ++c) {
      float xv = xo[r * 32 + c];
      q += xv * wq[d * 32 + c];
      k += xv * wk[d * 32 + c];
      v += xv * wv[d * 32 + c];
    }
    Q[e] = q; Km[e] = k; Vm[e] = v;
  }
  __syncthreads();

  {                                           // scores / sqrt(64)
    int qr = tid >> 3, kr = tid & 7;
    float s = 0.f;
    for (int d = 0; d < 64; ++d) s += Q[qr * 64 + d] * Km[kr * 64 + d];
    attn[tid] = s * 0.125f;
  }
  __syncthreads();

  if (tid < 8) {                              // row softmax
    float mx = -INFINITY;
    for (int k = 0; k < 8; ++k) mx = fmaxf(mx, attn[tid * 8 + k]);
    float s = 0.f;
    for (int k = 0; k < 8; ++k) {
      float e = expf(attn[tid * 8 + k] - mx);
      attn[tid * 8 + k] = e;
      s += e;
    }
    float inv = 1.f / s;
    for (int k = 0; k < 8; ++k) attn[tid * 8 + k] *= inv;
  }
  __syncthreads();

  {                                           // pooled = mean_q(attn @ V)
    float p = 0.f;
    for (int q = 0; q < 8; ++q)
      for (int k = 0; k < 8; ++k) p += attn[q * 8 + k] * Vm[k * 64 + tid];
    pooled[tid] = p * 0.125f;
  }
  __syncthreads();

  {                                           // out = pooled @ wo^T + bo
    float o = bo[tid];
    for (int e = 0; e < 64; ++e) o += pooled[e] * wo[tid * 64 + e];
    out[b * 64 + tid] = o;
  }
}

// -------------------------------------------------------------------------
extern "C" void kernel_launch(void* const* d_in, const int* in_sizes, int n_in,
                              void* d_out, int out_size, void* d_ws, size_t ws_size,
                              hipStream_t stream) {
  (void)in_sizes; (void)n_in; (void)out_size; (void)ws_size;
  const float* x    = (const float*)d_in[0];
  const float* w1   = (const float*)d_in[1];
  const float* b1   = (const float*)d_in[2];
  const float* g1   = (const float*)d_in[3];
  const float* bt1  = (const float*)d_in[4];
  const float* m1   = (const float*)d_in[5];
  const float* v1   = (const float*)d_in[6];
  const float* wr1  = (const float*)d_in[7];
  const float* br1  = (const float*)d_in[8];
  const float* wd1  = (const float*)d_in[9];
  const float* bd1  = (const float*)d_in[10];
  const float* w2   = (const float*)d_in[11];
  const float* b2   = (const float*)d_in[12];
  const float* g2   = (const float*)d_in[13];
  const float* bt2  = (const float*)d_in[14];
  const float* m2   = (const float*)d_in[15];
  const float* v2   = (const float*)d_in[16];
  const float* wr2  = (const float*)d_in[17];
  const float* br2  = (const float*)d_in[18];
  const float* wd2  = (const float*)d_in[19];
  const float* bd2  = (const float*)d_in[20];
  const float* wr3  = (const float*)d_in[21];
  const float* br3  = (const float*)d_in[22];
  const float* wf   = (const float*)d_in[23];
  const float* bf   = (const float*)d_in[24];
  const float* caw1 = (const float*)d_in[25];
  const float* caw2 = (const float*)d_in[26];
  const float* wq   = (const float*)d_in[27];
  const float* bq   = (const float*)d_in[28];
  const float* wk   = (const float*)d_in[29];
  const float* bk   = (const float*)d_in[30];
  const float* wv   = (const float*)d_in[31];
  const float* bv   = (const float*)d_in[32];
  const float* wo   = (const float*)d_in[33];
  const float* bo   = (const float*)d_in[34];

  _Float16* wf16 = (_Float16*)d_ws;
  float* fusedG  = (float*)((char*)d_ws + FUSED_BYTE_OFF);

  auto prep = [&](const float* src, int offs, int rows, int realRows, int Kreal, int Kpad) {
    int total = rows * Kpad;
    prep_weights<<<(total + 255) / 256, 256, 0, stream>>>(src, wf16 + offs, rows, realRows,
                                                          Kreal, Kpad);
  };
  prep(w1,  OFF_W1,  64,  64,  66,  96);
  prep(wr1, OFF_WR1, 64,  64,  22,  32);
  prep(wd1, OFF_WD1, 64,  64,  192, 192);
  prep(wr2, OFF_WR2, 128, 128, 64,  64);
  prep(w2,  OFF_W2,  128, 128, 192, 192);
  prep(wd2, OFF_WD2, 32,  32,  384, 384);
  prep(wr3, OFF_WR3, 32,  32,  128, 128);
  prep(wf,  OFF_WF,  32,  22,  32,  32);

  encode_kernel<<<64, NTHR, 0, stream>>>(x, wf16, fusedG,
                                         b1, g1, bt1, m1, v1, br1, bd1,
                                         b2, g2, bt2, m2, v2, br2, bd2, br3, bf,
                                         caw1, caw2);
  attn_kernel<<<8, 64, 0, stream>>>(fusedG, wq, bq, wk, bk, wv, bv, wo, bo, (float*)d_out);
}
